// BigramLanguageModel_29575144800539
// MI455X (gfx1250) — matrix operand order repaired and checked by reading, not compiled, not run on hardware
//
#include <hip/hip_runtime.h>
#include <cstdint>
#include <cstddef>

// ---------------------------------------------------------------------------
// Problem constants (match reference)
// ---------------------------------------------------------------------------
constexpr int Bc  = 4;
constexpr int Tc  = 1024;
constexpr int Ec  = 256;
constexpr int Hc  = 4;
constexpr int Lc  = 4;
constexpr int Vc  = 32000;
constexpr int HDc = 64;          // head dim
constexpr int BTc = Bc * Tc;     // 4096 token rows

typedef __attribute__((ext_vector_type(16))) __bf16 v16bf;
typedef __attribute__((ext_vector_type(8)))  float  v8f;

// ---------------------------------------------------------------------------
// WMMA helpers (CDNA5 wave32, 16x16x32 bf16 -> f32)
// ---------------------------------------------------------------------------
__device__ inline v8f wmma_bf16(v16bf a, v16bf b, v8f c) {
  // 8 args: (neg_a, A, neg_b, B, c_mod, C, reuse_a, reuse_b)
  return __builtin_amdgcn_wmma_f32_16x16x32_bf16(false, a, false, b,
                                                 (short)0, c, false, false);
}

// A-fragment (16x32 bf16, row-major source, leading dim = ld elements).
// ISA layout: lane half hf holds, for row M = lane&15:
//   elements 0..7  -> K = k0 + hf*8 + e        (contiguous 16B)
//   elements 8..15 -> K = k0 + 16 + hf*8 + e-8 (contiguous 16B)
__device__ inline v16bf load_a_frag(const __bf16* __restrict__ A, int ld,
                                    int row0, int k0, int lane) {
  const int hf = lane >> 4, r = lane & 15;
  const __bf16* p = A + (size_t)(row0 + r) * ld + k0 + hf * 8;
  union { v16bf v; uint4 q[2]; } u;
  u.q[0] = *(const uint4*)(p);
  u.q[1] = *(const uint4*)(p + 16);
  return u.v;
}

// B-fragment (32x16 bf16). Source is B^T, i.e. [N][K] row-major: lane half hf,
// col N = col0 + (lane&15), elements 0..7 -> K = k0 + hf*8 + e,
// elements 8..15 -> K = k0 + 16 + hf*8 + e-8.
__device__ inline v16bf load_b_frag(const __bf16* __restrict__ BT, int ld,
                                    int col0, int k0, int lane) {
  const int hf = lane >> 4, n = lane & 15;
  const __bf16* p = BT + (size_t)(col0 + n) * ld + k0 + hf * 8;
  union { v16bf v; uint4 q[2]; } u;
  u.q[0] = *(const uint4*)(p);
  u.q[1] = *(const uint4*)(p + 16);
  return u.v;
}

// ---------------------------------------------------------------------------
// Cross-lane reductions via DPP16 row_ror ring all-reduce (pure VALU, no LDS
// round-trip like ds_bpermute-based __shfl_xor).
// ---------------------------------------------------------------------------
template <int N>
__device__ inline float row_ror(float v) {
  return __int_as_float(__builtin_amdgcn_update_dpp(
      0, __float_as_int(v), 0x120 + N, 0xF, 0xF, true));
}
__device__ inline float grp16_max(float v) {   // all-reduce in 16-lane half
  v = fmaxf(v, row_ror<1>(v));
  v = fmaxf(v, row_ror<2>(v));
  v = fmaxf(v, row_ror<4>(v));
  v = fmaxf(v, row_ror<8>(v));
  return v;
}
__device__ inline float grp16_sum(float v) {
  v += row_ror<1>(v);
  v += row_ror<2>(v);
  v += row_ror<4>(v);
  v += row_ror<8>(v);
  return v;
}
__device__ inline float warp_sum32(float v) {
  v = grp16_sum(v);                 // per-half totals (DPP)
  v += __shfl_xor(v, 16, 32);       // single cross-half exchange
  return v;
}

// ---------------------------------------------------------------------------
// Weight convert + transpose: WT[n*K + k] = (bf16) W[k*N + n]
// ---------------------------------------------------------------------------
__global__ void convt_kernel(const float* __restrict__ W,
                             __bf16* __restrict__ WT, int Kd, int Nd) {
  size_t i = (size_t)blockIdx.x * blockDim.x + threadIdx.x;
  if (i >= (size_t)Kd * Nd) return;
  int n = (int)(i / Kd);
  int k = (int)(i % Kd);
  WT[i] = (__bf16)W[(size_t)k * Nd + n];
}

// ---------------------------------------------------------------------------
// Embedding: x[bt, c] = tok_emb[idx[bt], c] + pos_emb[bt % T, c]
// ---------------------------------------------------------------------------
__global__ void embed_kernel(const int* __restrict__ idx,
                             const float* __restrict__ tok,
                             const float* __restrict__ pos,
                             float* __restrict__ X) {
  size_t i = (size_t)blockIdx.x * blockDim.x + threadIdx.x;
  if (i >= (size_t)BTc * Ec) return;
  int bt = (int)(i / Ec);
  int c  = (int)(i % Ec);
  int t  = bt % Tc;
  X[i] = tok[(size_t)idx[bt] * Ec + c] + pos[(size_t)t * Ec + c];
}

// ---------------------------------------------------------------------------
// LayerNorm over E=256, one wave per row, bf16 output (feeds WMMA GEMMs)
// ---------------------------------------------------------------------------
__global__ __launch_bounds__(256) void ln_kernel(const float* __restrict__ X,
                                                 const float* __restrict__ g,
                                                 const float* __restrict__ bt,
                                                 __bf16* __restrict__ Y,
                                                 int Mrows) {
  const int lane = threadIdx.x & 31;
  const int wid  = threadIdx.x >> 5;
  const int row  = blockIdx.x * 8 + wid;
  if (row >= Mrows) return;
  const float* xp = X + (size_t)row * Ec + lane * 8;
  float a[8];
  {
    float4 v0 = *(const float4*)(xp);
    float4 v1 = *(const float4*)(xp + 4);
    a[0] = v0.x; a[1] = v0.y; a[2] = v0.z; a[3] = v0.w;
    a[4] = v1.x; a[5] = v1.y; a[6] = v1.z; a[7] = v1.w;
  }
  float s = 0.f;
#pragma unroll
  for (int i = 0; i < 8; ++i) s += a[i];
  s = warp_sum32(s);
  const float mean = s * (1.f / Ec);
  float ss = 0.f;
#pragma unroll
  for (int i = 0; i < 8; ++i) { a[i] -= mean; ss += a[i] * a[i]; }
  ss = warp_sum32(ss);
  const float rs = rsqrtf(ss * (1.f / Ec) + 1e-5f);
  __bf16* yp = Y + (size_t)row * Ec + lane * 8;
#pragma unroll
  for (int i = 0; i < 8; ++i) {
    int c = lane * 8 + i;
    yp[i] = (__bf16)(a[i] * rs * g[c] + bt[c]);
  }
}

// ---------------------------------------------------------------------------
// Generic WMMA GEMM: C[M,N] = A[M,K](bf16) * B[K,N] (given as BT=[N,K] bf16)
// Block = 8 waves (2 x 4); wave tile = 32 x (16*WN); block tile 64 x (64*WN).
// K compile-time (KT); k-loop software double-buffered.
// Epilogue flags: OB bf16 out, HB +bias[N], HR +fp32 residual, RL relu,
// OVT = scatter output transposed per head as VT[B][H][HD][T] (bf16), which
// makes the store a single 16B vector per tile (t is contiguous along e).
// Requires: M%64==0, N%(64*WN)==0, KT%32==0.
// ---------------------------------------------------------------------------
template <int OB, int HB, int HR, int RL, int OVT, int WN, int KT>
__global__ __launch_bounds__(256) void wmma_gemm_kernel(
    const __bf16* __restrict__ A, const __bf16* __restrict__ BT,
    const float* __restrict__ bias, const float* __restrict__ Rres,
    void* __restrict__ Cout, int N) {
  const int lane = threadIdx.x & 31;
  const int wave = threadIdx.x >> 5;
  const int M0 = blockIdx.y * 64 + (wave >> 2) * 32;
  const int N0 = (blockIdx.x * 4 + (wave & 3)) * (16 * WN);

  v8f acc[2][WN] = {};
  v16bf ca[2], cb[WN], na[2], nb[WN];
#pragma unroll
  for (int i = 0; i < 2; ++i) ca[i] = load_a_frag(A, KT, M0 + 16 * i, 0, lane);
#pragma unroll
  for (int j = 0; j < WN; ++j) cb[j] = load_b_frag(BT, KT, N0 + 16 * j, 0, lane);

#pragma unroll 4
  for (int k0 = 0; k0 < KT; k0 += 32) {
    const int kn = k0 + 32;
    if (kn < KT) {
#pragma unroll
      for (int i = 0; i < 2; ++i)
        na[i] = load_a_frag(A, KT, M0 + 16 * i, kn, lane);
#pragma unroll
      for (int j = 0; j < WN; ++j)
        nb[j] = load_b_frag(BT, KT, N0 + 16 * j, kn, lane);
      if (kn + 32 < KT)  // pull the B stream ahead into L2/L0
        __builtin_prefetch(BT + (size_t)(N0 + (lane & 15)) * KT + kn + 32, 0, 3);
    }
#pragma unroll
    for (int i = 0; i < 2; ++i)
#pragma unroll
      for (int j = 0; j < WN; ++j)
        acc[i][j] = wmma_bf16(ca[i], cb[j], acc[i][j]);
    if (kn < KT) {
#pragma unroll
      for (int i = 0; i < 2; ++i) ca[i] = na[i];
#pragma unroll
      for (int j = 0; j < WN; ++j) cb[j] = nb[j];
    }
  }

  const int hf = lane >> 4, nn = lane & 15;
#pragma unroll
  for (int i = 0; i < 2; ++i)
#pragma unroll
    for (int j = 0; j < WN; ++j) {
      if (OVT) {
        // VT[B][H][HD][T] scatter: within a tile, t runs contiguously over e.
        const int row0 = M0 + i * 16 + 8 * hf;       // first t of the 8
        const int col  = N0 + j * 16 + nn;
        const int bb = row0 >> 10;                   // row / Tc
        const int tt = row0 & (Tc - 1);
        const int dh = col >> 6;                     // head
        const int dd = col & (HDc - 1);              // dim within head
        union { uint4 q; __bf16 e8[8]; } st;
#pragma unroll
        for (int e = 0; e < 8; ++e) st.e8[e] = (__bf16)acc[i][j][e];
        *(uint4*)((__bf16*)Cout +
                  (((size_t)(bb * Hc + dh) * HDc + dd) * Tc + tt)) = st.q;
      } else {
#pragma unroll
        for (int e = 0; e < 8; ++e) {
          const int row = M0 + i * 16 + e + 8 * hf;  // C layout: M = e + 8*half
          const int col = N0 + j * 16 + nn;          //           N = lane&15
          float v = acc[i][j][e];
          if (HB) v += bias[col];
          if (RL) v = fmaxf(v, 0.f);
          if (HR) v += Rres[(size_t)row * N + col];
          if (OB) ((__bf16*)Cout)[(size_t)row * N + col] = (__bf16)v;
          else    ((float*)Cout)[(size_t)row * N + col] = v;
        }
      }
    }
}

// ---------------------------------------------------------------------------
// Flash attention, one wave per (b, h, 16 query rows).
// Q,K are bf16 [B,T,E] (head h in cols [h*64,h*64+64)); V is pre-transposed
// VT[B][H][HD][T] so P*V B-fragments are contiguous b128 loads.
// S = Q K^T via WMMA; online softmax in C-register layout with DPP16
// reductions; P staged through a 16x32 LDS tile (same-wave LDS ops are
// in-order, so no barrier needed in a single-wave workgroup); O += P V.
// ---------------------------------------------------------------------------
__global__ __launch_bounds__(32) void attn_kernel(const __bf16* __restrict__ Q,
                                                  const __bf16* __restrict__ Kb,
                                                  const __bf16* __restrict__ VT,
                                                  __bf16* __restrict__ O) {
  __shared__ float stage[16][33];   // 16 rows x 32 probs (+1 pad)
  const int lane = threadIdx.x;
  const int hf = lane >> 4, nn = lane & 15;
  const int qt = blockIdx.x;
  const int bh = blockIdx.y;
  const int b  = bh / Hc;
  const int hd = bh % Hc;
  const int qbase = qt * 16;
  const int ncols = qbase + 16;          // causal: keys [0, ncols)
  const float scale = 0.125f;            // 1/sqrt(64)

  const __bf16* qp  = Q + ((size_t)(b * Tc + qbase)) * Ec + hd * HDc;
  const __bf16* vtp = VT + (size_t)(b * Hc + hd) * HDc * Tc;  // [d][t]
  const v16bf aq0 = load_a_frag(qp, Ec, 0, 0, lane);
  const v16bf aq1 = load_a_frag(qp, Ec, 0, 32, lane);

  v8f co[4] = {};
  float m[8], l[8];
#pragma unroll
  for (int e = 0; e < 8; ++e) { m[e] = -1e30f; l[e] = 0.f; }

  for (int s0 = 0; s0 < ncols; s0 += 32) {
    const __bf16* kp = Kb + ((size_t)(b * Tc + s0)) * Ec + hd * HDc;

    // ---- S tiles: distinct fragment registers so all loads fly together --
    const v16bf bk00 = load_b_frag(kp, Ec, 0, 0, lane);
    const v16bf bk01 = load_b_frag(kp, Ec, 0, 32, lane);
    const v16bf bk10 = load_b_frag(kp, Ec, 16, 0, lane);
    const v16bf bk11 = load_b_frag(kp, Ec, 16, 32, lane);
    v8f st0 = {}, st1 = {};
    st0 = wmma_bf16(aq0, bk00, st0);
    st0 = wmma_bf16(aq1, bk01, st0);
    st1 = wmma_bf16(aq0, bk10, st1);
    st1 = wmma_bf16(aq1, bk11, st1);

    // ---- V fragments: contiguous b128 loads from VT, issued early -------
    const v16bf bv0 = load_b_frag(vtp, Tc, 0,  s0, lane);
    const v16bf bv1 = load_b_frag(vtp, Tc, 16, s0, lane);
    const v16bf bv2 = load_b_frag(vtp, Tc, 32, s0, lane);
    const v16bf bv3 = load_b_frag(vtp, Tc, 48, s0, lane);

    // ---- scale + causal mask + online softmax (DPP reductions) ----------
    float p0[8], p1[8];
#pragma unroll
    for (int e = 0; e < 8; ++e) {
      const int trow = qbase + e + 8 * hf;
      float v0 = st0[e] * scale; if (s0 + nn > trow)      v0 = -1e30f;
      float v1 = st1[e] * scale; if (s0 + 16 + nn > trow) v1 = -1e30f;
      const float mb = grp16_max(fmaxf(v0, v1));
      const float mn = fmaxf(m[e], mb);
      const float al = __expf(m[e] - mn);
      v0 = __expf(v0 - mn);
      v1 = __expf(v1 - mn);
      const float rs = grp16_sum(v0 + v1);
      l[e] = l[e] * al + rs;
      m[e] = mn;
      co[0][e] *= al; co[1][e] *= al; co[2][e] *= al; co[3][e] *= al;
      p0[e] = v0; p1[e] = v1;
    }

    // ---- stage P (C layout -> LDS); same-wave LDS is in-order -----------
#pragma unroll
    for (int e = 0; e < 8; ++e) {
      stage[e + 8 * hf][nn]      = p0[e];
      stage[e + 8 * hf][16 + nn] = p1[e];
    }

    // ---- A fragment of P from LDS (A layout) ----------------------------
    union { v16bf v; __bf16 h[16]; } ap;
#pragma unroll
    for (int e = 0; e < 16; ++e) {
      const int sidx = (e < 8) ? (hf * 8 + e) : (16 + hf * 8 + (e - 8));
      ap.h[e] = (__bf16)stage[nn][sidx];
    }

    // ---- O += P * V (4 N-tiles of head dim) -----------------------------
    co[0] = wmma_bf16(ap.v, bv0, co[0]);
    co[1] = wmma_bf16(ap.v, bv1, co[1]);
    co[2] = wmma_bf16(ap.v, bv2, co[2]);
    co[3] = wmma_bf16(ap.v, bv3, co[3]);
  }

  // ---- normalize + store (bf16 [B,T,E]) ---------------------------------
#pragma unroll
  for (int nt = 0; nt < 4; ++nt)
#pragma unroll
    for (int e = 0; e < 8; ++e) {
      const int trow = qbase + e + 8 * hf;
      O[((size_t)(b * Tc + trow)) * Ec + hd * HDc + nt * 16 + nn] =
          (__bf16)(co[nt][e] / l[e]);
    }
}

// ---------------------------------------------------------------------------
// Loss: single-pass online log-sum-exp per row (one read of the 524MB
// logits instead of two), then deterministic reduction.
// ---------------------------------------------------------------------------
__global__ __launch_bounds__(256) void rowloss_kernel(
    const float* __restrict__ logits, const int* __restrict__ tg,
    float* __restrict__ rl) {
  __shared__ float mred[256], sred[256];
  const int row = blockIdx.x;
  const int tid = threadIdx.x;
  const float* lp = logits + (size_t)row * Vc;
  float mloc = -1e30f, sloc = 0.f;
  for (int c = tid; c < Vc; c += 256) {
    const float v = lp[c];
    const float mn = fmaxf(mloc, v);
    sloc = sloc * __expf(mloc - mn) + __expf(v - mn);
    mloc = mn;
  }
  mred[tid] = mloc; sred[tid] = sloc; __syncthreads();
  for (int s = 128; s; s >>= 1) {
    if (tid < s) {
      const float m2 = mred[tid + s], s2 = sred[tid + s];
      const float mn = fmaxf(mred[tid], m2);
      sred[tid] = sred[tid] * __expf(mred[tid] - mn) + s2 * __expf(m2 - mn);
      mred[tid] = mn;
    }
    __syncthreads();
  }
  if (tid == 0) rl[row] = logf(sred[0]) + mred[0] - lp[tg[row]];
}

__global__ __launch_bounds__(256) void lossred_kernel(
    const float* __restrict__ rl, float* __restrict__ out) {
  __shared__ float red[256];
  const int tid = threadIdx.x;
  float s = 0.f;
  for (int i = tid; i < BTc; i += 256) s += rl[i];
  red[tid] = s; __syncthreads();
  for (int d = 128; d; d >>= 1) {
    if (tid < d) red[tid] += red[tid + d];
    __syncthreads();
  }
  if (tid == 0) out[0] = red[0] * (1.f / BTc);
}

// ---------------------------------------------------------------------------
// Host orchestration
// ---------------------------------------------------------------------------
extern "C" void kernel_launch(void* const* d_in, const int* in_sizes, int n_in,
                              void* d_out, int out_size, void* d_ws,
                              size_t ws_size, hipStream_t stream) {
  (void)in_sizes; (void)n_in; (void)out_size; (void)ws_size;
  const int*   idx     = (const int*)  d_in[0];
  const int*   targets = (const int*)  d_in[1];
  const float* tok_emb = (const float*)d_in[2];
  const float* pos_emb = (const float*)d_in[3];
  const float* Wq      = (const float*)d_in[4];
  const float* Wk      = (const float*)d_in[5];
  const float* Wv      = (const float*)d_in[6];
  const float* Wproj   = (const float*)d_in[7];
  const float* bproj   = (const float*)d_in[8];
  const float* W1      = (const float*)d_in[9];
  const float* b1      = (const float*)d_in[10];
  const float* W2      = (const float*)d_in[11];
  const float* b2      = (const float*)d_in[12];
  const float* ln1_g   = (const float*)d_in[13];
  const float* ln1_b   = (const float*)d_in[14];
  const float* ln2_g   = (const float*)d_in[15];
  const float* ln2_b   = (const float*)d_in[16];
  const float* lnf_g   = (const float*)d_in[17];
  const float* lnf_b   = (const float*)d_in[18];
  const float* Wout    = (const float*)d_in[19];
  const float* bout    = (const float*)d_in[20];

  float* logits   = (float*)d_out;
  float* loss_out = logits + (size_t)BTc * Vc;

  // Workspace bump allocator (~41 MB total)
  char* wp = (char*)d_ws;
  auto alloc = [&](size_t bytes) -> void* {
    void* p = wp;
    wp += (bytes + 255) & ~(size_t)255;
    return p;
  };
  float*  x     = (float*) alloc((size_t)BTc * Ec * 4);
  __bf16* h     = (__bf16*)alloc((size_t)BTc * Ec * 2);
  __bf16* qb    = (__bf16*)alloc((size_t)BTc * Ec * 2);
  __bf16* kb    = (__bf16*)alloc((size_t)BTc * Ec * 2);
  __bf16* vt    = (__bf16*)alloc((size_t)BTc * Ec * 2);   // VT[B][H][HD][T]
  __bf16* ao    = (__bf16*)alloc((size_t)BTc * Ec * 2);
  __bf16* mid   = (__bf16*)alloc((size_t)BTc * 4 * Ec * 2);
  __bf16* wqT   = (__bf16*)alloc((size_t)Ec * Ec * 2);
  __bf16* wkT   = (__bf16*)alloc((size_t)Ec * Ec * 2);
  __bf16* wvT   = (__bf16*)alloc((size_t)Ec * Ec * 2);
  __bf16* wpT   = (__bf16*)alloc((size_t)Ec * Ec * 2);
  __bf16* w1T   = (__bf16*)alloc((size_t)4 * Ec * Ec * 2);
  __bf16* w2T   = (__bf16*)alloc((size_t)4 * Ec * Ec * 2);
  __bf16* woutT = (__bf16*)alloc((size_t)Vc * Ec * 2);
  float*  rl    = (float*) alloc((size_t)BTc * 4);

  const dim3 blk256(256);
  auto cblocks = [](size_t n) { return dim3((unsigned)((n + 255) / 256)); };

  // x = tok_emb[idx] + pos_emb
  embed_kernel<<<cblocks((size_t)BTc * Ec), blk256, 0, stream>>>(
      idx, tok_emb, pos_emb, x);

  for (int l = 0; l < Lc; ++l) {
    // ---- attention block -------------------------------------------------
    ln_kernel<<<BTc / 8, blk256, 0, stream>>>(x, ln1_g + l * Ec,
                                              ln1_b + l * Ec, h, BTc);
    convt_kernel<<<cblocks((size_t)Ec * Ec), blk256, 0, stream>>>(
        Wq + (size_t)l * Ec * Ec, wqT, Ec, Ec);
    convt_kernel<<<cblocks((size_t)Ec * Ec), blk256, 0, stream>>>(
        Wk + (size_t)l * Ec * Ec, wkT, Ec, Ec);
    convt_kernel<<<cblocks((size_t)Ec * Ec), blk256, 0, stream>>>(
        Wv + (size_t)l * Ec * Ec, wvT, Ec, Ec);
    // q/k = h @ W -> bf16 [B,T,E];  v -> transposed VT[B][H][HD][T]
    wmma_gemm_kernel<1, 0, 0, 0, 0, 2, Ec>
        <<<dim3(Ec / 128, BTc / 64), blk256, 0, stream>>>(
            h, wqT, nullptr, nullptr, qb, Ec);
    wmma_gemm_kernel<1, 0, 0, 0, 0, 2, Ec>
        <<<dim3(Ec / 128, BTc / 64), blk256, 0, stream>>>(
            h, wkT, nullptr, nullptr, kb, Ec);
    wmma_gemm_kernel<1, 0, 0, 0, 1, 2, Ec>
        <<<dim3(Ec / 128, BTc / 64), blk256, 0, stream>>>(
            h, wvT, nullptr, nullptr, vt, Ec);
    attn_kernel<<<dim3(Tc / 16, Bc * Hc), dim3(32), 0, stream>>>(qb, kb, vt, ao);
    convt_kernel<<<cblocks((size_t)Ec * Ec), blk256, 0, stream>>>(
        Wproj + (size_t)l * Ec * Ec, wpT, Ec, Ec);
    // x = x + ao @ Wproj + bproj   (fp32 residual stream)
    wmma_gemm_kernel<0, 1, 1, 0, 0, 2, Ec>
        <<<dim3(Ec / 128, BTc / 64), blk256, 0, stream>>>(
            ao, wpT, bproj + (size_t)l * Ec, x, x, Ec);

    // ---- FFN block -------------------------------------------------------
    ln_kernel<<<BTc / 8, blk256, 0, stream>>>(x, ln2_g + l * Ec,
                                              ln2_b + l * Ec, h, BTc);
    convt_kernel<<<cblocks((size_t)Ec * 4 * Ec), blk256, 0, stream>>>(
        W1 + (size_t)l * Ec * 4 * Ec, w1T, Ec, 4 * Ec);
    // mid = relu(h @ W1 + b1) -> bf16  (WN=4: 64x256 block tile)
    wmma_gemm_kernel<1, 1, 0, 1, 0, 4, Ec>
        <<<dim3(4 * Ec / 256, BTc / 64), blk256, 0, stream>>>(
            h, w1T, b1 + (size_t)l * 4 * Ec, nullptr, mid, 4 * Ec);
    convt_kernel<<<cblocks((size_t)4 * Ec * Ec), blk256, 0, stream>>>(
        W2 + (size_t)l * 4 * Ec * Ec, w2T, 4 * Ec, Ec);
    // x = x + mid @ W2 + b2   (K = 1024)
    wmma_gemm_kernel<0, 1, 1, 0, 0, 2, 4 * Ec>
        <<<dim3(Ec / 128, BTc / 64), blk256, 0, stream>>>(
            mid, w2T, b2 + (size_t)l * Ec, x, x, Ec);
  }

  // ---- final LN + classifier + loss -------------------------------------
  ln_kernel<<<BTc / 8, blk256, 0, stream>>>(x, lnf_g, lnf_b, h, BTc);
  convt_kernel<<<cblocks((size_t)Ec * Vc), blk256, 0, stream>>>(Wout, woutT,
                                                                Ec, Vc);
  wmma_gemm_kernel<0, 1, 0, 0, 0, 4, Ec>
      <<<dim3(Vc / 256, BTc / 64), blk256, 0, stream>>>(
          h, woutT, bout, nullptr, logits, Vc);
  rowloss_kernel<<<BTc, blk256, 0, stream>>>(logits, targets, rl);
  lossred_kernel<<<1, blk256, 0, stream>>>(rl, loss_out);
}
